// xLSTMStack_71073118814363
// MI455X (gfx1250) — compile-verified, matrix operand order checked by reading, not run on hardware
//
#include <hip/hip_runtime.h>
#include <hip/hip_bf16.h>
#include <math.h>

// ---------------------------------------------------------------------------
// Dimensions (fixed by the reference)
// ---------------------------------------------------------------------------
#define BB    16
#define LL    2048
#define INDIM 64
#define HH    512
#define H3    1536
#define NLAY  4

typedef __attribute__((ext_vector_type(16))) _Float16 v16h;
typedef __attribute__((ext_vector_type(8)))  float    v8f;

#if __has_builtin(__builtin_amdgcn_global_load_async_to_lds_b128) && \
    __has_builtin(__builtin_amdgcn_s_wait_asynccnt)
#define HAVE_ASYNC_LDS 1
typedef int v4i_ __attribute__((vector_size(16)));
#define AS_G_V4I(p) ((__attribute__((address_space(1))) v4i_*)(p))
#define AS_L_V4I(p) ((__attribute__((address_space(3))) v4i_*)(p))
#endif

// ---------------------------------------------------------------------------
// WMMA accumulator: one wave computes one 16x16 f32 tile of
//   C = A(f32)[M,K] @ W(f32)[N,K]^T
// via v_wmma_f32_16x16x32_f16 (operands converted to f16 fragments).
// Fragment packing per CDNA5 ISA 7.12.2 (16-bit A 16x32 / B 32x16):
//   lanes 0-15:  elems 0..7 -> K=k0..k0+7,   8..15 -> K=k0+16..k0+23
//   lanes 16-31: elems 0..7 -> K=k0+8..k0+15, 8..15 -> K=k0+24..k0+31
// ---------------------------------------------------------------------------
__device__ __forceinline__ v8f wmma_acc_f32(const float* __restrict__ A,
                                            const float* __restrict__ W,
                                            int m0, int n0, int K) {
  const int lane = threadIdx.x & 31;
  const int kb   = (lane & 16) ? 8 : 0;
  const float* ap0 = A + (size_t)(m0 + (lane & 15)) * K + kb;
  const float* wp0 = W + (size_t)(n0 + (lane & 15)) * K + kb;
  v8f acc = {};
#pragma unroll 4
  for (int kt = 0; kt < K; kt += 32) {
    v16h a, b;
    const float* ap = ap0 + kt;
    const float* wp = wp0 + kt;
#pragma unroll
    for (int j = 0; j < 8; ++j) {
      a[j]     = (_Float16)ap[j];
      a[j + 8] = (_Float16)ap[j + 16];
      b[j]     = (_Float16)wp[j];
      b[j + 8] = (_Float16)wp[j + 16];
    }
    acc = __builtin_amdgcn_wmma_f32_16x16x32_f16(false, a, false, b,
                                                 (short)0, acc, false, false);
  }
  return acc;
}

// ---------------------------------------------------------------------------
// Input projection: h = x @ Wp^T + bp   (M=B*L=32768, N=512, K=64)
// grid(256, 32) x 256 threads; wave w handles m-tile blockIdx.x*8+w
// ---------------------------------------------------------------------------
__global__ __launch_bounds__(256) void proj_kernel(const float* __restrict__ x,
                                                   const float* __restrict__ Wp,
                                                   const float* __restrict__ bp,
                                                   float* __restrict__ h) {
  const int m0 = (blockIdx.x * 8 + (threadIdx.x >> 5)) * 16;
  const int n0 = blockIdx.y * 16;
  v8f acc = wmma_acc_f32(x, Wp, m0, n0, INDIM);
  const int lane  = threadIdx.x & 31;
  const int ncol  = n0 + (lane & 15);
  const int mbase = m0 + ((lane & 16) ? 8 : 0);
  const float bv  = bp[ncol];
#pragma unroll
  for (int r = 0; r < 8; ++r)
    h[(size_t)(mbase + r) * HH + ncol] = acc[r] + bv;
}

// ---------------------------------------------------------------------------
// xg = hmix @ Wih^T + bih  (M=32768, N=1536, K=512), stored f16
// grid(256, 96) x 256 threads
// ---------------------------------------------------------------------------
__global__ __launch_bounds__(256) void xg_kernel(const float* __restrict__ hmix,
                                                 const float* __restrict__ Wih,
                                                 const float* __restrict__ bih,
                                                 _Float16* __restrict__ xg) {
  const int m0 = (blockIdx.x * 8 + (threadIdx.x >> 5)) * 16;
  const int n0 = blockIdx.y * 16;
  v8f acc = wmma_acc_f32(hmix, Wih, m0, n0, HH);
  const int lane  = threadIdx.x & 31;
  const int ncol  = n0 + (lane & 15);
  const int mbase = m0 + ((lane & 16) ? 8 : 0);
  const float bv  = bih[ncol];
#pragma unroll
  for (int r = 0; r < 8; ++r)
    xg[(size_t)(mbase + r) * H3 + ncol] = (_Float16)(acc[r] + bv);
}

// ---------------------------------------------------------------------------
// LayerNorm per row of 512.  grid = B*L blocks of 256.
// ---------------------------------------------------------------------------
__global__ __launch_bounds__(256) void ln_kernel(const float* __restrict__ h,
                                                 const float* __restrict__ g,
                                                 const float* __restrict__ b,
                                                 float* __restrict__ out) {
  __shared__ float red[256];
  const size_t row = blockIdx.x;
  const float* x = h + row * HH;
  float s = 0.f;
  for (int i = threadIdx.x; i < HH; i += 256) s += x[i];
  red[threadIdx.x] = s; __syncthreads();
  for (int o = 128; o > 0; o >>= 1) { if (threadIdx.x < o) red[threadIdx.x] += red[threadIdx.x + o]; __syncthreads(); }
  const float mu = red[0] / (float)HH; __syncthreads();
  float v = 0.f;
  for (int i = threadIdx.x; i < HH; i += 256) { float d = x[i] - mu; v += d * d; }
  red[threadIdx.x] = v; __syncthreads();
  for (int o = 128; o > 0; o >>= 1) { if (threadIdx.x < o) red[threadIdx.x] += red[threadIdx.x + o]; __syncthreads(); }
  const float inv = rsqrtf(red[0] / (float)HH + 1e-5f);
  for (int i = threadIdx.x; i < HH; i += 256)
    out[row * HH + i] = (x[i] - mu) * inv * g[i] + b[i];
}

// ---------------------------------------------------------------------------
// Fused depthwise-conv(3 branches) + BN(eval) + GELU + softmax gate + mix.
// One block per (batch b, 16-step L tile). Dynamic LDS:
//   tile  [16+22][512] f32  (input with halo, zero padded)
//   outs  [1536][16]   f32  (all branch activations for this tile)
// ---------------------------------------------------------------------------
#define LT  16
#define PAD 11
__global__ __launch_bounds__(256) void conv_gate_kernel(
    const float* __restrict__ hln,
    const float* __restrict__ cw0, const float* __restrict__ cb0,
    const float* __restrict__ cw1, const float* __restrict__ cb1,
    const float* __restrict__ cw2, const float* __restrict__ cb2,
    const float* __restrict__ bng, const float* __restrict__ bnb,
    const float* __restrict__ gW,  const float* __restrict__ gb,
    float* __restrict__ hmix) {
  extern __shared__ float smem[];
  float* tile = smem;                         // (LT+2*PAD)*HH
  float* outs = tile + (LT + 2 * PAD) * HH;   // H3*LT
  __shared__ float logits[LT][3];
  __shared__ float wgt[LT][3];

  const int b  = blockIdx.x;
  const int l0 = blockIdx.y * LT;

  for (int idx = threadIdx.x; idx < (LT + 2 * PAD) * HH; idx += 256) {
    int ll = idx / HH, c = idx % HH;
    int l = l0 + ll - PAD;
    tile[idx] = (l >= 0 && l < LL) ? hln[((size_t)b * LL + l) * HH + c] : 0.0f;
  }
  __syncthreads();

  const float bn_scale = rsqrtf(1.0f + 1e-5f);
  for (int job = threadIdx.x; job < H3; job += 256) {
    const int i = job >> 9, c = job & (HH - 1);
    const float* w; const float* cb; int k;
    if      (i == 0) { w = cw0 + c * 5;  cb = cb0; k = 5;  }
    else if (i == 1) { w = cw1 + c * 11; cb = cb1; k = 11; }
    else             { w = cw2 + c * 23; cb = cb2; k = 23; }
    const int half = k >> 1;
    const float bias = cb[c];
    const float gsc = bng[i * HH + c] * bn_scale, gbi = bnb[i * HH + c];
    for (int l = 0; l < LT; ++l) {
      float s = bias;
      int base = (l + PAD - half) * HH + c;
      for (int t = 0; t < k; ++t) s += tile[base + t * HH] * w[t];
      s = s * gsc + gbi;
      outs[job * LT + l] = 0.5f * s * (1.0f + erff(s * 0.70710678118654752f));
    }
  }
  __syncthreads();

  // gate logits: wave handles 2 timesteps; lanes stride the 1536 channels
  const int wave = threadIdx.x >> 5, lane = threadIdx.x & 31;
  for (int li = 0; li < 2; ++li) {
    const int l = wave * 2 + li;
    float a0 = 0.f, a1 = 0.f, a2 = 0.f;
    for (int ch = lane; ch < H3; ch += 32) {
      float v = outs[ch * LT + l];
      a0 += v * gW[ch];
      a1 += v * gW[H3 + ch];
      a2 += v * gW[2 * H3 + ch];
    }
    for (int o = 16; o > 0; o >>= 1) {
      a0 += __shfl_down(a0, o);
      a1 += __shfl_down(a1, o);
      a2 += __shfl_down(a2, o);
    }
    if (lane == 0) { logits[l][0] = a0 + gb[0]; logits[l][1] = a1 + gb[1]; logits[l][2] = a2 + gb[2]; }
  }
  __syncthreads();
  if (threadIdx.x < LT) {
    const int l = threadIdx.x;
    float a = logits[l][0], bq = logits[l][1], cq = logits[l][2];
    float m = fmaxf(a, fmaxf(bq, cq));
    float e0 = expf(a - m), e1 = expf(bq - m), e2 = expf(cq - m);
    float inv = 1.0f / (e0 + e1 + e2);
    wgt[l][0] = e0 * inv; wgt[l][1] = e1 * inv; wgt[l][2] = e2 * inv;
  }
  __syncthreads();
  for (int idx = threadIdx.x; idx < LT * HH; idx += 256) {
    int l = idx >> 9, c = idx & (HH - 1);
    float v = outs[c * LT + l] * wgt[l][0]
            + outs[(HH + c) * LT + l] * wgt[l][1]
            + outs[(2 * HH + c) * LT + l] * wgt[l][2];
    hmix[((size_t)b * LL + l0 + l) * HH + c] = v;
  }
}

// ---------------------------------------------------------------------------
// Persistent GRU scan. 8 blocks x 384 threads (12 waves). Each block owns 64
// hidden columns -> 192 gate columns (r,z,n) = 12 WMMA N-tiles, one per wave.
// Whh slice (f16, 192KB) lives in LDS for the whole scan; per step only the
// 16x512 f16 hidden state moves (global ping-pong via L2, pulled into LDS
// with GLOBAL_LOAD_ASYNC_TO_LDS when available) + one atomic grid barrier.
// ---------------------------------------------------------------------------
#define NBLK 8
__global__ __launch_bounds__(384) void gru_scan_kernel(
    const _Float16* __restrict__ xg,     // [B,L,3H] f16
    const float*    __restrict__ Whh,    // [3H,512]
    const float*    __restrict__ bhh,    // [3H]
    const float*    __restrict__ hmix,   // [B,L,H]
    _Float16*       __restrict__ hstate, // [2][16][512] f16 (pre-zeroed)
    int*            __restrict__ counter,
    float*          __restrict__ hout) {
  extern __shared__ char smemraw[];
  _Float16* whh = (_Float16*)smemraw;        // [192][512]
  _Float16* hp  = whh + 192 * HH;            // [16][512]
  float*    hg  = (float*)(hp + BB * HH);    // [192][16]

  const int tid = threadIdx.x;
  const int lane = tid & 31, wave = tid >> 5;
  const int c0 = blockIdx.x * 64;
  const int kb = (lane & 16) ? 8 : 0;

  // Preload Whh slice (rows {g*512 + c0 + cl} for g=0..2, cl=0..63) as f16.
  for (int idx = tid; idx < 192 * HH; idx += 384) {
    int nloc = idx >> 9, k = idx & (HH - 1);
    int grow = (nloc >> 6) * HH + c0 + (nloc & 63);
    whh[idx] = (_Float16)Whh[(size_t)grow * HH + k];
  }
  __syncthreads();

  for (int t = 0; t < LL; ++t) {
    const _Float16* hprev = hstate + (size_t)(t & 1) * BB * HH;
    _Float16*       hnext = hstate + (size_t)((t + 1) & 1) * BB * HH;

    // hidden state (16KB f16) global -> LDS
#ifdef HAVE_ASYNC_LDS
    for (int idx = tid * 8; idx < BB * HH; idx += 384 * 8)
      __builtin_amdgcn_global_load_async_to_lds_b128(AS_G_V4I(hprev + idx),
                                                     AS_L_V4I(hp + idx), 0, 0);
    __builtin_amdgcn_s_wait_asynccnt(0);
#else
    for (int idx = tid; idx < BB * HH; idx += 384)
      hp[idx] = ((const volatile _Float16*)hprev)[idx];
#endif
    __syncthreads();

    // hg = h_prev @ Whh_slice^T : 12 waves x (16x16x512), operands in LDS
    {
      const int nloc = wave * 16 + (lane & 15);
      const int m    = lane & 15;
      const _Float16* ap0 = hp  + m * HH + kb;
      const _Float16* wp0 = whh + nloc * HH + kb;
      v8f acc = {};
#pragma unroll
      for (int kt = 0; kt < HH; kt += 32) {
        v16h a, b;
        const _Float16* ap = ap0 + kt;
        const _Float16* wp = wp0 + kt;
#pragma unroll
        for (int j = 0; j < 8; ++j) {
          a[j] = ap[j]; a[j + 8] = ap[j + 16];
          b[j] = wp[j]; b[j + 8] = wp[j + 16];
        }
        acc = __builtin_amdgcn_wmma_f32_16x16x32_f16(false, a, false, b,
                                                     (short)0, acc, false, false);
      }
      const int ncol = wave * 16 + (lane & 15);
      const int mb   = (lane & 16) ? 8 : 0;
#pragma unroll
      for (int r = 0; r < 8; ++r) hg[ncol * 16 + mb + r] = acc[r];
    }
    __syncthreads();

    // gates for (b, col) in this block's 16x64 slice
    for (int e = tid; e < BB * 64; e += 384) {
      const int b = e & 15, cl = e >> 4, col = c0 + cl;
      float hr = hg[cl * 16 + b]          + bhh[col];
      float hz = hg[(64 + cl) * 16 + b]   + bhh[HH + col];
      float hn = hg[(128 + cl) * 16 + b]  + bhh[2 * HH + col];
      const _Float16* xrow = xg + ((size_t)b * LL + t) * H3;
      float xr = (float)xrow[col];
      float xz = (float)xrow[HH + col];
      float xn = (float)xrow[2 * HH + col];
      float hpv = (float)hp[b * HH + col];
      float r = 1.0f / (1.0f + expf(-(xr + hr)));
      float z = 1.0f / (1.0f + expf(-(xz + hz)));
      float n = tanhf(xn + r * hn);
      float hnew = (1.0f - z) * n + z * hpv;
      hnext[b * HH + col] = (_Float16)hnew;
      const size_t orow = ((size_t)b * LL + t) * HH + col;
      hout[orow] = hmix[orow] + hnew;
      if (t + 1 < LL) __builtin_prefetch(xrow + H3 + col, 0, 1); // global_prefetch
    }

    // grid barrier (ping-pong makes one barrier per step sufficient)
    __syncthreads();
    if (tid == 0) {
      __threadfence();
      atomicAdd(counter, 1);
      while (*(volatile int*)counter < NBLK * (t + 1)) __builtin_amdgcn_s_sleep(1);
      __threadfence();
    }
    __syncthreads();
  }
}

// ---------------------------------------------------------------------------
// Host-side orchestration. Workspace layout (bytes):
//   [0, 64MB)        hln   f32 [B,L,H]
//   [64MB, 128MB)    hmix  f32 [B,L,H]
//   [128MB, 224MB)   xg    f16 [B,L,3H]
//   [+0, +32KB)      hstate f16 [2][16][512]
//   [+32KB, +32KB+64) counter
// ---------------------------------------------------------------------------
extern "C" void kernel_launch(void* const* d_in, const int* in_sizes, int n_in,
                              void* d_out, int out_size, void* d_ws, size_t ws_size,
                              hipStream_t stream) {
  (void)in_sizes; (void)n_in; (void)out_size; (void)ws_size;
  const float* x    = (const float*)d_in[0];
  const float* Wp   = (const float*)d_in[1];
  const float* bp   = (const float*)d_in[2];
  const float* ln_g = (const float*)d_in[3];
  const float* ln_b = (const float*)d_in[4];
  const float* cw0  = (const float*)d_in[5];
  const float* cb0  = (const float*)d_in[6];
  const float* cw1  = (const float*)d_in[7];
  const float* cb1  = (const float*)d_in[8];
  const float* cw2  = (const float*)d_in[9];
  const float* cb2  = (const float*)d_in[10];
  const float* bng  = (const float*)d_in[11];
  const float* bnb  = (const float*)d_in[12];
  const float* gW   = (const float*)d_in[13];
  const float* gb   = (const float*)d_in[14];
  const float* Wih  = (const float*)d_in[15];
  const float* Whh  = (const float*)d_in[16];
  const float* bih  = (const float*)d_in[17];
  const float* bhh  = (const float*)d_in[18];

  float* h = (float*)d_out;
  char* ws = (char*)d_ws;
  const size_t ACT = (size_t)BB * LL * HH * sizeof(float);        // 64MB
  float*    hln     = (float*)ws;
  float*    hmix    = (float*)(ws + ACT);
  _Float16* xg      = (_Float16*)(ws + 2 * ACT);
  _Float16* hstate  = (_Float16*)(ws + 2 * ACT + (size_t)BB * LL * H3 * sizeof(_Float16));
  int*      counter = (int*)((char*)hstate + 2 * BB * HH * sizeof(_Float16));

  // input projection
  proj_kernel<<<dim3(BB * LL / 128, HH / 16), 256, 0, stream>>>(x, Wp, bp, h);

  const size_t convShm = ((size_t)(LT + 2 * PAD) * HH + (size_t)H3 * LT) * sizeof(float);
  const size_t scanShm = (size_t)192 * HH * 2 + (size_t)BB * HH * 2 + (size_t)192 * 16 * 4;

  for (int l = 0; l < NLAY; ++l) {
    ln_kernel<<<BB * LL, 256, 0, stream>>>(h, ln_g + l * HH, ln_b + l * HH, hln);
    conv_gate_kernel<<<dim3(BB, LL / LT), 256, convShm, stream>>>(
        hln,
        cw0 + (size_t)l * HH * 5,  cb0 + (size_t)l * HH,
        cw1 + (size_t)l * HH * 11, cb1 + (size_t)l * HH,
        cw2 + (size_t)l * HH * 23, cb2 + (size_t)l * HH,
        bng + (size_t)l * 3 * HH,  bnb + (size_t)l * 3 * HH,
        gW  + (size_t)l * 3 * H3,  gb + (size_t)l * 3,
        hmix);
    xg_kernel<<<dim3(BB * LL / 128, H3 / 16), 256, 0, stream>>>(
        hmix, Wih + (size_t)l * H3 * HH, bih + (size_t)l * H3, xg);
    (void)hipMemsetAsync(hstate, 0, 2 * BB * HH * sizeof(_Float16) + 64, stream);
    gru_scan_kernel<<<NBLK, 384, scanShm, stream>>>(
        xg, Whh + (size_t)l * H3 * HH, bhh + (size_t)l * H3, hmix,
        hstate, counter, h);
  }
}
